// MultiHeadAttention_24721831756536
// MI455X (gfx1250) — matrix operand order repaired and checked by reading, not compiled, not run on hardware
//
#include <hip/hip_runtime.h>
#include <hip/hip_bf16.h>

typedef __attribute__((ext_vector_type(16))) __bf16 v16bf;
typedef __attribute__((ext_vector_type(8)))  __bf16 v8bf;
typedef __attribute__((ext_vector_type(8)))  float  v8f;

#define NSEQ 4096
#define DM   768
#define NH   12
#define DKH  64
#define LDP  72   // padded LDS row stride (elements) for the 32x64 weight panel

// ---------------- WMMA helper (bf16 16x16x32, f32 accumulate) ----------------

__device__ __forceinline__ v8f wmma_bf16(v16bf a, v16bf b, v8f c) {
  return __builtin_amdgcn_wmma_f32_16x16x32_bf16(false, a, false, b, (short)0, c,
                                                 false, false);
}

// A operand: 16x32 bf16, row-major, leading dim `ld` (elements).
// lane<16 -> row=lane,    K in {0..7} U {16..23}
// lane>=16 -> row=lane-16, K in {8..15} U {24..31}
__device__ __forceinline__ v16bf load_a16(const __bf16* tile, int ld, int lane) {
  const int r  = lane & 15;
  const int kh = (lane >> 4) & 1;
  const __bf16* p = tile + (size_t)r * ld + kh * 8;
  v8bf lo = *(const v8bf*)(p);
  v8bf hi = *(const v8bf*)(p + 16);
  v16bf o;
#pragma unroll
  for (int i = 0; i < 8; ++i) { o[i] = lo[i]; o[8 + i] = hi[i]; }
  return o;
}

// B operand: 32x16 bf16 (KxN), row-major, leading dim `ld`. col = lane&15, K as in load_a16.
__device__ __forceinline__ v16bf load_b16(const __bf16* tile, int ld, int lane) {
  const int n  = lane & 15;
  const int kh = (lane >> 4) & 1;
  const __bf16* p = tile + n + (size_t)(kh * 8) * ld;
  v16bf o;
#pragma unroll
  for (int i = 0; i < 8; ++i) { o[i] = p[(size_t)i * ld]; o[8 + i] = p[(size_t)(16 + i) * ld]; }
  return o;
}

// ---------------- gfx1250 async copy to LDS (ASYNCcnt path) ----------------
// GV mode: each lane moves 16B from its own global address to its own LDS address.

__device__ __forceinline__ void async_stage_b128(unsigned lds_off, const void* g) {
  asm volatile("global_load_async_to_lds_b128 %0, %1, off"
               :: "v"(lds_off), "v"(g)
               : "memory");
}
__device__ __forceinline__ void wait_async0() {
  asm volatile("s_wait_asynccnt 0" ::: "memory");
}

// Stage a 32(K) x 64(N) bf16 panel of Wt (global ld = DM) into an LDS buffer
// with padded row stride LDP. 128 threads: thread t covers row=t>>2, 16-col
// chunk ch=t&3, i.e. 32 bytes => two async b128 ops per thread (one per wave
// instruction; addresses are per-lane).
__device__ __forceinline__ void stage_panel(__bf16* lbuf, const __bf16* gsrc, int tid) {
  const int row = tid >> 2;
  const int ch  = tid & 3;
  const __bf16* g = gsrc + (size_t)row * DM + ch * 16;
  unsigned l = (unsigned)(size_t)(lbuf + row * LDP + ch * 16);
  async_stage_b128(l,      g);
  async_stage_b128(l + 16, g + 8);
}

// ---------------- Prep kernels ----------------

__global__ void cvt_bf16_kernel(const float* __restrict__ in,
                                __bf16* __restrict__ out, int n) {
  int i = blockIdx.x * blockDim.x + threadIdx.x;
  if (i < n) out[i] = (__bf16)in[i];
}

// W [out][in] (torch Linear) -> Wt [in][out] bf16.
__global__ void transpose_w_kernel(const float* __restrict__ W,
                                   __bf16* __restrict__ Wt) {
  int i = blockIdx.x * blockDim.x + threadIdx.x;
  if (i >= DM * DM) return;
  int o = i / DM, k = i % DM;
  Wt[(size_t)k * DM + o] = (__bf16)W[(size_t)o * DM + k];
}

// ---------------- Projection GEMM with async-staged, double-buffered W panel ----
// Block = 128 threads (4 waves). Block computes 64(M) x 64(N); wave w handles
// rows [mtbase + 16w, +16). grid = (NSEQ/64, NH). N-tile == one head.
// transposed==0: out[h][m][dk]   (q, v)
// transposed==1: out[h][dk][m]   (k stored transposed for the score GEMM)

__global__ void proj_kernel(const __bf16* __restrict__ X,
                            const __bf16* __restrict__ Wt,
                            const float* __restrict__ bias,
                            __bf16* __restrict__ out, int transposed) {
  const int tid  = threadIdx.x;
  const int lane = tid & 31;
  const int wave = tid >> 5;
  const int h  = blockIdx.y;
  const int mtbase = blockIdx.x * 64;

  __shared__ __align__(16) __bf16 bpan[2][32 * LDP];

  v8f acc[4];
#pragma unroll
  for (int t = 0; t < 4; ++t)
#pragma unroll
    for (int i = 0; i < 8; ++i) acc[t][i] = 0.f;

  const __bf16* a_base = X + (size_t)(mtbase + wave * 16) * DM;
  const __bf16* b_base = Wt + (size_t)h * DKH;

  stage_panel(bpan[0], b_base, tid);
  wait_async0();
  __syncthreads();

  for (int kb = 0; kb < DM; kb += 32) {
    const int cur = (kb >> 5) & 1;
    if (kb + 32 < DM)
      stage_panel(bpan[cur ^ 1], b_base + (size_t)(kb + 32) * DM, tid);

    if (kb + 64 < DM) __builtin_prefetch(a_base + kb + 64, 0, 0);
    v16bf a = load_a16(a_base + kb, DM, lane);
#pragma unroll
    for (int t = 0; t < 4; ++t) {
      v16bf b = load_b16(&bpan[cur][t * 16], LDP, lane);
      acc[t] = wmma_bf16(a, b, acc[t]);
    }

    wait_async0();     // next panel landed (this wave's share)
    __syncthreads();   // all waves staged & done reading current panel
  }

  const int col = lane & 15;
  const int rh  = (lane >> 4) * 8;  // C layout: row = i + 8*(lane>=16)
#pragma unroll
  for (int t = 0; t < 4; ++t) {
    float bv = bias[h * DKH + t * 16 + col];
#pragma unroll
    for (int i = 0; i < 8; ++i) {
      float val = acc[t][i] + bv;
      int m   = mtbase + wave * 16 + rh + i;
      int dkk = t * 16 + col;
      if (!transposed)
        out[((size_t)h * NSEQ + m) * DKH + dkk] = (__bf16)val;
      else
        out[((size_t)h * DKH + dkk) * NSEQ + m] = (__bf16)val;
    }
  }
}

// ---------------- Flash attention: one wave per (16-query tile, head) ----------------

__global__ void attn_kernel(const __bf16* __restrict__ qh,
                            const __bf16* __restrict__ kth,
                            const __bf16* __restrict__ vh,
                            __bf16* __restrict__ out /* [n][768] */) {
  const int mt = blockIdx.x;
  const int h  = blockIdx.y;
  const int lane = threadIdx.x;

  const __bf16* q  = qh  + ((size_t)h * NSEQ + mt * 16) * DKH;
  const __bf16* kt = kth + (size_t)h * DKH * NSEQ;   // [64][4096]
  const __bf16* v  = vh  + (size_t)h * NSEQ * DKH;   // [4096][64]

  v16bf a0 = load_a16(q,      DKH, lane);
  v16bf a1 = load_a16(q + 32, DKH, lane);

  v8f oacc[4];
  float mrow[8], lrow[8];
#pragma unroll
  for (int i = 0; i < 8; ++i) { mrow[i] = -3.0e38f; lrow[i] = 0.f; }
#pragma unroll
  for (int t = 0; t < 4; ++t)
#pragma unroll
    for (int i = 0; i < 8; ++i) oacc[t][i] = 0.f;

  __shared__ __align__(32) __bf16 p_lds[16 * 32];
  const float scale = 0.125f;  // 1/sqrt(64)
  const int col = lane & 15;
  const int rh  = (lane >> 4) * 8;

  for (int kb = 0; kb < NSEQ; kb += 32) {
    if (kb + 32 < NSEQ) {
      __builtin_prefetch(v + (size_t)(kb + 32) * DKH, 0, 0);
      __builtin_prefetch(kt + kb + 32, 0, 0);
    }

    // ---- S = q . k^T : 16 queries x 32 keys
    v8f s0, s1;
#pragma unroll
    for (int i = 0; i < 8; ++i) { s0[i] = 0.f; s1[i] = 0.f; }
#pragma unroll
    for (int c = 0; c < 2; ++c) {  // dk chunks of 32
      v16bf aa = c ? a1 : a0;
      v16bf b0 = load_b16(kt + (size_t)(c * 32) * NSEQ + kb,      NSEQ, lane);
      v16bf b1 = load_b16(kt + (size_t)(c * 32) * NSEQ + kb + 16, NSEQ, lane);
      s0 = wmma_bf16(aa, b0, s0);
      s1 = wmma_bf16(aa, b1, s1);
    }

    // ---- online softmax (rows live across 16 lanes of each half-wave)
#pragma unroll
    for (int i = 0; i < 8; ++i) {
      float x0 = s0[i] * scale, x1 = s1[i] * scale;
      float mx = fmaxf(x0, x1);
#pragma unroll
      for (int off = 8; off >= 1; off >>= 1)
        mx = fmaxf(mx, __shfl_xor(mx, off, 16));
      float mnew  = fmaxf(mrow[i], mx);
      float alpha = __expf(mrow[i] - mnew);
      mrow[i] = mnew;
      float p0 = __expf(x0 - mnew);
      float p1 = __expf(x1 - mnew);
      s0[i] = p0; s1[i] = p1;
      float ps = p0 + p1;
#pragma unroll
      for (int off = 8; off >= 1; off >>= 1)
        ps += __shfl_xor(ps, off, 16);
      lrow[i] = lrow[i] * alpha + ps;
#pragma unroll
      for (int t = 0; t < 4; ++t) oacc[t][i] *= alpha;
    }

    // ---- C-layout -> A-layout for P via LDS (16x32 bf16 tile)
#pragma unroll
    for (int i = 0; i < 8; ++i) {
      p_lds[(rh + i) * 32 + col]      = (__bf16)s0[i];
      p_lds[(rh + i) * 32 + col + 16] = (__bf16)s1[i];
    }
    __threadfence_block();
    v16bf pa = load_a16(p_lds, 32, lane);
    __threadfence_block();

    // ---- O += P @ V  (K=32 keys, N=64 dk -> 4 WMMAs)
#pragma unroll
    for (int t = 0; t < 4; ++t) {
      v16bf bv = load_b16(v + (size_t)kb * DKH + t * 16, DKH, lane);
      oacc[t] = wmma_bf16(pa, bv, oacc[t]);
    }
  }

  // ---- normalize, emit concatenated-head layout [n][768] bf16
#pragma unroll
  for (int t = 0; t < 4; ++t) {
#pragma unroll
    for (int i = 0; i < 8; ++i) {
      int m = mt * 16 + rh + i;
      float val = oacc[t][i] / lrow[i];
      out[(size_t)m * DM + h * DKH + t * 16 + col] = (__bf16)val;
    }
  }
}

// ---------------- Output projection (same staged-panel scheme, f32 out) ----------------

__global__ void oproj_kernel(const __bf16* __restrict__ X,
                             const __bf16* __restrict__ Wt,
                             const float* __restrict__ bias,
                             float* __restrict__ out) {
  const int tid  = threadIdx.x;
  const int lane = tid & 31;
  const int wave = tid >> 5;
  const int nt = blockIdx.y;        // 64-wide N tile
  const int mtbase = blockIdx.x * 64;

  __shared__ __align__(16) __bf16 bpan[2][32 * LDP];

  v8f acc[4];
#pragma unroll
  for (int t = 0; t < 4; ++t)
#pragma unroll
    for (int i = 0; i < 8; ++i) acc[t][i] = 0.f;

  const __bf16* a_base = X + (size_t)(mtbase + wave * 16) * DM;
  const __bf16* b_base = Wt + (size_t)nt * 64;

  stage_panel(bpan[0], b_base, tid);
  wait_async0();
  __syncthreads();

  for (int kb = 0; kb < DM; kb += 32) {
    const int cur = (kb >> 5) & 1;
    if (kb + 32 < DM)
      stage_panel(bpan[cur ^ 1], b_base + (size_t)(kb + 32) * DM, tid);

    if (kb + 64 < DM) __builtin_prefetch(a_base + kb + 64, 0, 0);
    v16bf a = load_a16(a_base + kb, DM, lane);
#pragma unroll
    for (int t = 0; t < 4; ++t) {
      v16bf b = load_b16(&bpan[cur][t * 16], LDP, lane);
      acc[t] = wmma_bf16(a, b, acc[t]);
    }

    wait_async0();
    __syncthreads();
  }

  const int col = lane & 15;
  const int rh  = (lane >> 4) * 8;
#pragma unroll
  for (int t = 0; t < 4; ++t) {
    float bv = bias[nt * 64 + t * 16 + col];
#pragma unroll
    for (int i = 0; i < 8; ++i) {
      int m = mtbase + wave * 16 + rh + i;
      out[(size_t)m * DM + nt * 64 + t * 16 + col] = acc[t][i] + bv;
    }
  }
}

// ---------------- Host launcher ----------------

extern "C" void kernel_launch(void* const* d_in, const int* in_sizes, int n_in,
                              void* d_out, int out_size, void* d_ws, size_t ws_size,
                              hipStream_t stream) {
  const float* Qf = (const float*)d_in[0];
  const float* Kf = (const float*)d_in[1];
  const float* Vf = (const float*)d_in[2];
  const float* Wq = (const float*)d_in[3];
  const float* bq = (const float*)d_in[4];
  const float* Wk = (const float*)d_in[5];
  const float* bk = (const float*)d_in[6];
  const float* Wv = (const float*)d_in[7];
  const float* bv = (const float*)d_in[8];
  const float* Wo = (const float*)d_in[9];
  const float* bo = (const float*)d_in[10];
  float* out = (float*)d_out;

  char* ws = (char*)d_ws;
  size_t off = 0;
  auto ALLOC = [&](size_t bytes) -> void* {
    void* p = ws + off;
    off += (bytes + 255) & ~(size_t)255;
    return p;
  };

  const size_t XB = (size_t)NSEQ * DM * sizeof(__bf16);
  const size_t WB = (size_t)DM * DM * sizeof(__bf16);
  const size_t HB = (size_t)NH * NSEQ * DKH * sizeof(__bf16);

  __bf16* Qb   = (__bf16*)ALLOC(XB);
  __bf16* Kb   = (__bf16*)ALLOC(XB);
  __bf16* Vb   = (__bf16*)ALLOC(XB);
  __bf16* WqT  = (__bf16*)ALLOC(WB);
  __bf16* WkT  = (__bf16*)ALLOC(WB);
  __bf16* WvT  = (__bf16*)ALLOC(WB);
  __bf16* WoT  = (__bf16*)ALLOC(WB);
  __bf16* qH   = (__bf16*)ALLOC(HB);  // [h][n][64]
  __bf16* kT   = (__bf16*)ALLOC(HB);  // [h][64][n]
  __bf16* vH   = (__bf16*)ALLOC(HB);  // [h][n][64]
  __bf16* attn = (__bf16*)ALLOC(XB);  // [n][768]

  const int nel = NSEQ * DM;
  cvt_bf16_kernel<<<(nel + 255) / 256, 256, 0, stream>>>(Qf, Qb, nel);
  cvt_bf16_kernel<<<(nel + 255) / 256, 256, 0, stream>>>(Kf, Kb, nel);
  cvt_bf16_kernel<<<(nel + 255) / 256, 256, 0, stream>>>(Vf, Vb, nel);

  const int wel = DM * DM;
  transpose_w_kernel<<<(wel + 255) / 256, 256, 0, stream>>>(Wq, WqT);
  transpose_w_kernel<<<(wel + 255) / 256, 256, 0, stream>>>(Wk, WkT);
  transpose_w_kernel<<<(wel + 255) / 256, 256, 0, stream>>>(Wv, WvT);
  transpose_w_kernel<<<(wel + 255) / 256, 256, 0, stream>>>(Wo, WoT);

  dim3 gp(NSEQ / 64, NH);  // 64 x 12 blocks, 128 threads each
  proj_kernel<<<gp, 128, 0, stream>>>(Qb, WqT, bq, qH, 0);
  proj_kernel<<<gp, 128, 0, stream>>>(Kb, WkT, bk, kT, 1);
  proj_kernel<<<gp, 128, 0, stream>>>(Vb, WvT, bv, vH, 0);

  dim3 ga(NSEQ / 16, NH);  // 256 x 12 single-wave blocks
  attn_kernel<<<ga, 32, 0, stream>>>(qH, kT, vH, attn);

  oproj_kernel<<<gp, 128, 0, stream>>>(attn, WoT, bo, out);
}
